// SimpleConvLSTM_6665789244071
// MI455X (gfx1250) — matrix operand order repaired and checked by reading, not compiled, not run on hardware
//
#include <hip/hip_runtime.h>
#include <hip/hip_bf16.h>
#include <math.h>

// ---------------- problem constants ----------------
#define BATCH   64
#define TSTEPS  20
#define CIN_X   25
#define HID     64
#define HW      625      // 25*25
#define NPIX    40000    // BATCH*HW
#define PADHW   729      // 27*27 zero-halo frame
#define CHSTR   46656    // BATCH*PADHW : stride between channels (halves)

typedef __attribute__((ext_vector_type(16))) _Float16 v16h;
typedef __attribute__((ext_vector_type(8)))  _Float16 v8h;
typedef __attribute__((ext_vector_type(8)))  float    v8f;

#define LDA 48   // halves per LDS row (96B: keeps every v16h frag 32B-aligned)
#define LDB 48

// =====================================================================
// Implicit-GEMM 3x3 SAME conv via WMMA f16->f32.
//   act : f16 [cin_pad][BATCH][27][27] (zero halo; pad channels are zero)
//   wk  : f16 [cout][9*cin_pad]        (K contiguous; K = tap*cin_pad + ci)
//   out : f32 [cout][NPIX]
// grid = (NPIX/64, cout/64), block = 128 threads (4 waves), wave tile 32x32.
// =====================================================================
__launch_bounds__(128)
__global__ void conv3x3_wmma(const _Float16* __restrict__ act,
                             const _Float16* __restrict__ wk,
                             float* __restrict__ out,
                             int cin_pad) {
  __shared__ __align__(128) _Float16 lA[64 * LDA];  // [m][k]  linear k
  __shared__ __align__(128) _Float16 lB[64 * LDB];  // [n][k]  linear k

  const int kstride = 9 * cin_pad;
  const int tid  = threadIdx.x;
  const int lane = tid & 31;
  const int wid  = tid >> 5;
  const int hi   = lane >> 4;        // lane group (K split)
  const int lo   = lane & 15;        // row within tile (M for A, N for B/D)
  const int wm   = (wid >> 1) << 5;  // wave M offset in block tile (0/32)
  const int wn   = (wid & 1)  << 5;  // wave N offset (0/32)

  const int nbase = blockIdx.x * 64;
  const int mbase = blockIdx.y * 64;

  // B-tile ownership: this thread owns pixel column p and 16 contiguous
  // channel rows starting at rbase.  One v16h LDS store per k-step.
  const int p     = tid & 63;
  const int rbase = (tid >> 6) << 4;       // 0 or 16
  {
  }
  const int n  = nbase + p;
  const int b  = n / HW;
  const int yx = n - b * HW;
  const int pbase = b * PADHW + (yx / 25 + 1) * 27 + (yx % 25 + 1);
  // per-thread activation base: channel rbase, center tap of pixel n
  const _Float16* actp = act + (size_t)rbase * CHSTR + pbase;
  _Float16* lBp = lB + p * LDB + rbase;

  // A-tile ownership: two 8-half vectors per thread.
  // q = tid + i*128 -> m = q>>2, koff = (q&3)*8, stored linearly at column koff.
  v8f acc[2][2];
#pragma unroll
  for (int mt = 0; mt < 2; ++mt)
#pragma unroll
    for (int nt = 0; nt < 2; ++nt) acc[mt][nt] = {};

  for (int tap = 0; tap < 9; ++tap) {
    const int shift = (tap / 3 - 1) * 27 + (tap % 3 - 1);
    for (int kc = 0; kc < cin_pad; kc += 32) {
      __syncthreads();
      // ---- A tile: 64 M-rows x 32 K halves, stored in linear k order ----
      // ISA A layout (16-bit 16x32): lane group hi holds K = {hi*8..+7, 16+hi*8..+7}
#pragma unroll
      for (int i = 0; i < 2; ++i) {
        int q     = tid + i * 128;          // 8-half vector id, 0..255
        int m     = q >> 2;
        int koff  = (q & 3) << 3;           // 0,8,16,24
        int sbase = koff; // 0,8,16,24
        v8h wv = *(const v8h*)(wk + (size_t)(mbase + m) * kstride
                                  + tap * cin_pad + kc + koff);
        *(v8h*)(lA + m * LDA + sbase) = wv;
      }
      // ---- B tile: 16 coalesced u16 loads -> one 32B LDS store ----
      {
        const _Float16* ap = actp + (size_t)kc * CHSTR + shift;
        if (kc + 32 < cin_pad)
          __builtin_prefetch(ap + (size_t)32 * CHSTR, 0, 1);
        v16h bvec;
#pragma unroll
        for (int i = 0; i < 16; ++i) bvec[i] = ap[(size_t)i * CHSTR];
        *(v16h*)lBp = bvec;
      }
      __syncthreads();
      // ---- fragments (one 32B LDS read each) + 4 WMMAs ----
      v16h afrag[2], bfrag[2];
#pragma unroll
      for (int mt = 0; mt < 2; ++mt)
        afrag[mt] = *(const v16h*)(lA + (wm + mt * 16 + lo) * LDA + hi * 16);
#pragma unroll
      for (int nt = 0; nt < 2; ++nt)
        bfrag[nt] = *(const v16h*)(lB + (wn + nt * 16 + lo) * LDB + hi * 16);
#pragma unroll
      for (int mt = 0; mt < 2; ++mt)
#pragma unroll
        for (int nt = 0; nt < 2; ++nt)
          acc[mt][nt] = __builtin_amdgcn_wmma_f32_16x16x32_f16(
              false, afrag[mt], false, bfrag[nt], (short)0, acc[mt][nt],
              false, false);
    }
  }

  // ---- store: C/D layout = lane lo -> N, vgpr r -> M = r + 8*hi ----
#pragma unroll
  for (int mt = 0; mt < 2; ++mt)
#pragma unroll
    for (int nt = 0; nt < 2; ++nt)
#pragma unroll
      for (int r = 0; r < 8; ++r) {
        int m = mbase + wm + mt * 16 + (hi << 3) + r;
        int nn = nbase + wn + nt * 16 + lo;
        out[(size_t)m * NPIX + nn] = acc[mt][nt][r];
      }
}

// ================= LayerNorm over C=25, pack f16 into padded frame =========
__global__ void ln_pack(const float* __restrict__ x,
                        const float* __restrict__ gamma,
                        const float* __restrict__ beta,
                        _Float16* __restrict__ dst, int t) {
  int n = blockIdx.x * blockDim.x + threadIdx.x;
  if (n >= NPIX) return;
  int b = n / HW, yx = n - b * HW;
  const float* xp = x + ((size_t)(b * TSTEPS + t) * CIN_X) * HW + yx;
  float v[CIN_X];
  float mu = 0.f;
#pragma unroll
  for (int c = 0; c < CIN_X; ++c) { v[c] = xp[c * HW]; mu += v[c]; }
  mu *= (1.f / CIN_X);
  float var = 0.f;
#pragma unroll
  for (int c = 0; c < CIN_X; ++c) { float d = v[c] - mu; var += d * d; }
  var *= (1.f / CIN_X);
  float rstd = rsqrtf(var + 1e-5f);
  int pb = b * PADHW + (yx / 25 + 1) * 27 + (yx % 25 + 1);
#pragma unroll
  for (int c = 0; c < CIN_X; ++c)
    dst[(size_t)c * CHSTR + pb] = (_Float16)((v[c] - mu) * rstd * gamma[c] + beta[c]);
}

// ================= Fused LSTM gate update =================================
// z: f32 [4*HID][NPIX] gate-major (i,f,o,g).  Writes c in place and h (f16)
// into up to two padded input frames (the channel-concat for the next convs).
__device__ __forceinline__ float sigmoidf_(float x) { return 1.f / (1.f + expf(-x)); }

__global__ void lstm_gate(const float* __restrict__ z,
                          const float* __restrict__ bias,
                          float* __restrict__ cbuf,
                          _Float16* __restrict__ dst1, int choff1,
                          _Float16* __restrict__ dst2, int choff2) {
  int idx = blockIdx.x * blockDim.x + threadIdx.x;
  if (idx >= HID * NPIX) return;
  int h = idx / NPIX, n = idx - h * NPIX;
  float zi = z[(size_t)(h)        * NPIX + n] + bias[h];
  float zf = z[(size_t)(HID + h)  * NPIX + n] + bias[HID + h];
  float zo = z[(size_t)(2*HID + h)* NPIX + n] + bias[2 * HID + h];
  float zg = z[(size_t)(3*HID + h)* NPIX + n] + bias[3 * HID + h];
  float cn = sigmoidf_(zf) * cbuf[idx] + sigmoidf_(zi) * tanhf(zg);
  cbuf[idx] = cn;
  float hv = sigmoidf_(zo) * tanhf(cn);
  int b = n / HW, yx = n - b * HW;
  int pb = b * PADHW + (yx / 25 + 1) * 27 + (yx % 25 + 1);
  _Float16 hh = (_Float16)hv;
  dst1[(size_t)(choff1 + h) * CHSTR + pb] = hh;
  if (dst2) dst2[(size_t)(choff2 + h) * CHSTR + pb] = hh;
}

// ================= weight repack: [O][I][3][3] f32 -> [O][9*Ipad] f16 ======
__global__ void wconv(const float* __restrict__ w, _Float16* __restrict__ wk,
                      int I, int Ipad, int O) {
  int idx = blockIdx.x * blockDim.x + threadIdx.x;
  int tot = O * 9 * Ipad;
  if (idx >= tot) return;
  int o = idx / (9 * Ipad);
  int rem = idx - o * 9 * Ipad;
  int tap = rem / Ipad;
  int i = rem - tap * Ipad;
  float v = (i < I) ? w[((size_t)o * I + i) * 9 + tap] : 0.f;
  wk[(size_t)o * (9 * Ipad) + tap * Ipad + i] = (_Float16)v;
}

// ================= head: bias + relu + 2x2 maxpool, torch-NCHW flatten =====
__global__ void relu_pool(const float* __restrict__ y, const float* __restrict__ bc,
                          float* __restrict__ pooled) {
  int idx = blockIdx.x * blockDim.x + threadIdx.x;
  if (idx >= BATCH * HID * 144) return;
  int b = idx / (HID * 144);
  int rem = idx - b * HID * 144;
  int c = rem / 144;
  int pyx = rem - c * 144;
  int py = pyx / 12, px = pyx - py * 12;
  const float* yp = y + (size_t)c * NPIX + b * HW;
  float m = yp[(2 * py) * 25 + 2 * px];
  m = fmaxf(m, yp[(2 * py) * 25 + 2 * px + 1]);
  m = fmaxf(m, yp[(2 * py + 1) * 25 + 2 * px]);
  m = fmaxf(m, yp[(2 * py + 1) * 25 + 2 * px + 1]);
  pooled[(size_t)b * 9216 + c * 144 + pyx] = fmaxf(m + bc[c], 0.f);
}

// ================= small FC layers =========================================
__global__ void fc(const float* __restrict__ in, const float* __restrict__ W,
                   const float* __restrict__ bias, float* __restrict__ out,
                   int B, int K, int O, int relu) {
  int idx = blockIdx.x * blockDim.x + threadIdx.x;
  if (idx >= B * O) return;
  int b = idx / O, o = idx - b * O;
  float s = bias[o];
  const float* ip = in + (size_t)b * K;
  const float* wp = W + (size_t)o * K;
  for (int k = 0; k < K; ++k) s += ip[k] * wp[k];
  out[idx] = relu ? fmaxf(s, 0.f) : s;
}

__global__ void fc3_lsm(const float* __restrict__ in, const float* __restrict__ W,
                        const float* __restrict__ bias, float* __restrict__ out) {
  int b = blockIdx.x * blockDim.x + threadIdx.x;
  if (b >= BATCH) return;
  float y0 = bias[0], y1 = bias[1];
  const float* ip = in + b * HID;
#pragma unroll
  for (int k = 0; k < HID; ++k) { y0 += ip[k] * W[k]; y1 += ip[k] * W[HID + k]; }
  float m = fmaxf(y0, y1);
  float lse = m + logf(expf(y0 - m) + expf(y1 - m));
  out[b * 2 + 0] = y0 - lse;
  out[b * 2 + 1] = y1 - lse;
}

// ================= zero-fill (graph-capture safe init) =====================
__global__ void zero_fill(uint4* __restrict__ p, size_t nvec) {
  size_t i = (size_t)blockIdx.x * blockDim.x + threadIdx.x;
  if (i < nvec) p[i] = uint4{0u, 0u, 0u, 0u};
}

// ===========================================================================
extern "C" void kernel_launch(void* const* d_in, const int* in_sizes, int n_in,
                              void* d_out, int out_size, void* d_ws, size_t ws_size,
                              hipStream_t stream) {
  (void)in_sizes; (void)n_in; (void)out_size; (void)ws_size;
  const float* x    = (const float*)d_in[0];
  const float* lng  = (const float*)d_in[1];
  const float* lnb  = (const float*)d_in[2];
  const float* w0   = (const float*)d_in[3];
  const float* b0   = (const float*)d_in[4];
  const float* w1   = (const float*)d_in[5];
  const float* b1   = (const float*)d_in[6];
  const float* wc1  = (const float*)d_in[7];
  const float* bc1  = (const float*)d_in[8];
  const float* wfc1 = (const float*)d_in[9];
  const float* bfc1 = (const float*)d_in[10];
  const float* wfc2 = (const float*)d_in[11];
  const float* bfc2 = (const float*)d_in[12];
  const float* wfc3 = (const float*)d_in[13];
  const float* bfc3 = (const float*)d_in[14];
  float* outp = (float*)d_out;

  char* ws = (char*)d_ws;
  size_t off = 0;
  auto alloc = [&](size_t bytes) -> char* {
    char* p = ws + off;
    off += (bytes + 255) & ~(size_t)255;
    return p;
  };

  // Zero-initialized region first (halos = SAME padding, h/c state,
  // and layer0's 7 pad channels which stay zero forever).
  _Float16* in0 = (_Float16*)alloc((size_t)96 * CHSTR * 2);        // [96][B][27][27]
  _Float16* in1 = (_Float16*)alloc((size_t)(2 * HID) * CHSTR * 2); // [128][B][27][27]
  float*    c0  = (float*)alloc((size_t)HID * NPIX * 4);
  float*    c1  = (float*)alloc((size_t)HID * NPIX * 4);
  size_t zero_bytes = off;
  float*    z   = (float*)alloc((size_t)4 * HID * NPIX * 4);       // gates / head out
  _Float16* wk0 = (_Float16*)alloc((size_t)256 * 9 * 96 * 2);
  _Float16* wk1 = (_Float16*)alloc((size_t)256 * 9 * 128 * 2);
  _Float16* wkc = (_Float16*)alloc((size_t)64 * 9 * 64 * 2);
  float* pooled = (float*)alloc((size_t)BATCH * 9216 * 4);
  float* f1o    = (float*)alloc((size_t)BATCH * 128 * 4);
  float* f2o    = (float*)alloc((size_t)BATCH * 64 * 4);

  size_t nvec = zero_bytes / 16;
  zero_fill<<<(unsigned)((nvec + 255) / 256), 256, 0, stream>>>((uint4*)ws, nvec);

  // Weight repack (deterministic, every call).
  wconv<<<(256 * 9 * 96  + 255) / 256, 256, 0, stream>>>(w0,  wk0, 89,  96,  256);
  wconv<<<(256 * 9 * 128 + 255) / 256, 256, 0, stream>>>(w1,  wk1, 128, 128, 256);
  wconv<<<(64  * 9 * 64  + 255) / 256, 256, 0, stream>>>(wc1, wkc, 64,  64,  64);

  const int gateBlocks = (HID * NPIX + 255) / 256;
  for (int t = 0; t < TSTEPS; ++t) {
    ln_pack<<<(NPIX + 255) / 256, 256, 0, stream>>>(x, lng, lnb, in0, t);
    // layer 0: in0 = [x(25) | h0(64) | 0(7)] -> gates z
    conv3x3_wmma<<<dim3(NPIX / 64, 4), 128, 0, stream>>>(in0, wk0, z, 96);
    // gate update: h0 -> in0 ch 25..88 (self) and in1 ch 0..63 (next layer)
    lstm_gate<<<gateBlocks, 256, 0, stream>>>(z, b0, c0, in0, CIN_X, in1, 0);
    // layer 1: in1 = [h0(64) | h1(64)] -> gates z
    conv3x3_wmma<<<dim3(NPIX / 64, 4), 128, 0, stream>>>(in1, wk1, z, 128);
    lstm_gate<<<gateBlocks, 256, 0, stream>>>(z, b1, c1, in1, HID, (_Float16*)nullptr, 0);
  }

  // head conv on final h1 (in1 channels 64..127), reuse z as [64][NPIX] output
  conv3x3_wmma<<<dim3(NPIX / 64, 1), 128, 0, stream>>>(in1 + (size_t)HID * CHSTR,
                                                       wkc, z, 64);
  relu_pool<<<(BATCH * HID * 144 + 255) / 256, 256, 0, stream>>>(z, bc1, pooled);
  fc<<<(BATCH * 128 + 255) / 256, 256, 0, stream>>>(pooled, wfc1, bfc1, f1o, BATCH, 9216, 128, 1);
  fc<<<(BATCH * 64  + 255) / 256, 256, 0, stream>>>(f1o,    wfc2, bfc2, f2o, BATCH, 128,  64, 1);
  fc3_lsm<<<1, 64, 0, stream>>>(f2o, wfc3, bfc3, outp);
}
